// Mamba2_32212254720184
// MI455X (gfx1250) — matrix operand order repaired and checked by reading, not compiled, not run on hardware
//
#include <hip/hip_runtime.h>

#define D_INNER 1024
#define CONV_DIM 1088
#define D_IN_PROJ 2240
#define LL 4096
#define BL 32768          // 8 * 4096

typedef __attribute__((ext_vector_type(16))) __bf16 v16bf;
typedef __attribute__((ext_vector_type(8)))  float  v8f;

union Frag {
    v16bf v;
    uint4 u4[2];
};

__device__ __forceinline__ unsigned short f2bf(float f) {
    unsigned int u = __float_as_uint(f);
    unsigned int r = u + 0x7FFFu + ((u >> 16) & 1u);   // round-to-nearest-even
    return (unsigned short)(r >> 16);
}
__device__ __forceinline__ float silu_f(float x) { return x / (1.f + __expf(-x)); }
__device__ __forceinline__ float softplus_f(float x) { return x > 20.f ? x : log1pf(__expf(x)); }

// CDNA5 async DMA: global -> LDS, 16B per lane, tracked by ASYNCcnt.
__device__ __forceinline__ void async_ld16(const void* gaddr, unsigned lds_off) {
    asm volatile("global_load_async_to_lds_b128 %0, %1, off"
                 :: "v"(lds_off), "v"(gaddr) : "memory");
}

// ---------------------------------------------------------------- fp32 -> bf16
__global__ void cvt_bf16(const float* __restrict__ in, unsigned short* __restrict__ out, int n) {
    int i = blockIdx.x * 256 + threadIdx.x;
    if (i < n) out[i] = f2bf(in[i]);
}

// ---------------------------------------------------------------- WMMA GEMM
// C[M,N] (f32, ldc) = A[M,K](bf16 row-major) * B[N,K](bf16 row-major)^T
// Block tile: 256(M) x 64(N), 8 waves; wave tile 32x64 = 2x4 WMMA 16x16 tiles.
// K chunks of 32, double-buffered in LDS, filled by GLOBAL_LOAD_ASYNC_TO_LDS_B128.
__global__ __launch_bounds__(256) void gemm_bf16_wmma(
    const unsigned short* __restrict__ A, const unsigned short* __restrict__ B,
    float* __restrict__ C, int K, int ldc)
{
    // uint4 units: bufA0 [0,1024) bufB0 [1024,1280) bufA1 [1280,2304) bufB1 [2304,2560)
    __shared__ uint4 lds[2560];     // 40 KB

    const int t    = threadIdx.x;
    const int wid  = t >> 5;
    const int lane = t & 31;
    const int ln   = lane & 15;
    const int half = lane >> 4;
    const long m0 = (long)blockIdx.x * 256;
    const long n0 = (long)blockIdx.y * 64;

    // Issue one K-chunk (32 columns) into buffer `buf`: 5 async 16B ops per thread.
    auto issue = [&](int kc, int buf) {
        const int aBase = buf ? 1280 : 0;
        const int bBase = buf ? 2304 : 1024;
#pragma unroll
        for (int j = 0; j < 4; ++j) {
            int c   = t + j * 256;            // 1024 x 16B chunks of A tile
            int row = c >> 2, grp = c & 3;
            async_ld16((const void*)(A + (m0 + row) * (long)K + kc + grp * 8),
                       (unsigned)(uintptr_t)&lds[aBase + c]);
        }
        {
            int row = t >> 2, grp = t & 3;    // 256 x 16B chunks of B tile
            async_ld16((const void*)(B + (n0 + row) * (long)K + kc + grp * 8),
                       (unsigned)(uintptr_t)&lds[bBase + t]);
        }
    };

    v8f acc[2][4] = {};
    const int NC = K >> 5;

    issue(0, 0);
    if (NC > 1) issue(32, 1);

    for (int i = 0; i < NC; ++i) {
        // Retire batch i (async completion is in-order; batch i+1 may stay in flight).
        if (i + 1 < NC) asm volatile("s_wait_asynccnt 5" ::: "memory");
        else            asm volatile("s_wait_asynccnt 0" ::: "memory");
        __syncthreads();

        const int aBase = (i & 1) ? 1280 : 0;
        const int bBase = (i & 1) ? 2304 : 1024;
        const uint4* lA = &lds[aBase];
        const uint4* lB = &lds[bBase];

        Frag af[2], bfr[4];
#pragma unroll
        for (int tm = 0; tm < 2; ++tm) {
            int r = wid * 32 + tm * 16 + ln;           // A: lane holds row r
            af[tm].u4[0] = lA[r * 4 + half];           // K 0..7  / 8..15
            af[tm].u4[1] = lA[r * 4 + 2 + half];       // K 16..23 / 24..31
        }
#pragma unroll
        for (int tn = 0; tn < 4; ++tn) {
            int n = tn * 16 + ln;                      // B: lane holds col n
            bfr[tn].u4[0] = lB[n * 4 + half];          // K 0..7  / 8..15
            bfr[tn].u4[1] = lB[n * 4 + 2 + half];      // K 16..23 / 24..31
        }
#pragma unroll
        for (int tm = 0; tm < 2; ++tm)
#pragma unroll
            for (int tn = 0; tn < 4; ++tn)
                acc[tm][tn] = __builtin_amdgcn_wmma_f32_16x16x32_bf16(
                    false, af[tm].v, false, bfr[tn].v, (short)0, acc[tm][tn], false, false);

        __syncthreads();                               // all waves done reading buf[i&1]
        if (i + 2 < NC) issue((i + 2) << 5, i & 1);    // refill the buffer just drained
    }

    // C/D layout: lane ln = col, VGPR v -> row v + 8*half
#pragma unroll
    for (int tm = 0; tm < 2; ++tm)
#pragma unroll
        for (int tn = 0; tn < 4; ++tn) {
            long n = n0 + tn * 16 + ln;
#pragma unroll
            for (int v = 0; v < 8; ++v) {
                long m = m0 + wid * 32 + tm * 16 + v + half * 8;
                C[m * (long)ldc + n] = acc[tm][tn][v];
            }
        }
}

// ---------------------------------------------------------------- dA = softplus(dt+bias)*exp(A_log)
__global__ void dA_kernel(const float* __restrict__ zx, const float* __restrict__ dt_bias,
                          const float* __restrict__ A_log, float* __restrict__ dA, int total) {
    int i = blockIdx.x * 256 + threadIdx.x;
    if (i >= total) return;
    int h   = i & 127;
    long bl = i >> 7;
    float v = zx[bl * (long)D_IN_PROJ + 2112 + h] + dt_bias[h];
    dA[i] = softplus_f(v) * __expf(A_log[h]);
}

// ---------------------------------------------------------------- fused depthwise conv stage -> xBCc
__global__ void conv_xbc_kernel(const float* __restrict__ zx, float* __restrict__ xBCc,
    const float* __restrict__ w33,
    const float* __restrict__ wv1, const float* __restrict__ wh1,   // w31x1, w13x1
    const float* __restrict__ wh2, const float* __restrict__ wv2,   // w13x2, w31x2
    const float* __restrict__ whb1, const float* __restrict__ wvb1, // w13bc1, w31bc1
    const float* __restrict__ whb2, const float* __restrict__ wvb2) // w13bc2, w31bc2
{
    long idx = (long)blockIdx.x * 256 + threadIdx.x;
    const long total = (long)BL * CONV_DIM;
    if (idx >= total) return;
    int j   = (int)(idx % CONV_DIM);
    long bl = idx / CONV_DIM;
    int b = (int)(bl >> 12);
    int l = (int)(bl & 4095);
    int h = l >> 6, w = l & 63;
    const float* base = zx + (long)b * LL * D_IN_PROJ;

    auto ld = [&](int hh, int ww, int src) -> float {
        if ((unsigned)hh >= 64u || (unsigned)ww >= 64u) return 0.f;
        return base[(long)(hh * 64 + ww) * D_IN_PROJ + 1024 + src];
    };

    float out = 0.f;
    if ((j & 1) == 0) {                         // even channels: 3x3 dwconv
        int c = j >> 1;
        int src = 2 * c;
        const float* wk = w33 + c * 9;
#pragma unroll
        for (int kh = 0; kh < 3; ++kh)
#pragma unroll
            for (int kw = 0; kw < 3; ++kw)
                out += wk[kh * 3 + kw] * ld(h + kh - 1, w + kw - 1, src);
    } else {                                    // odd channels: separable two-stage
        int c = (j - 1) >> 1;
        int src; const float *w1, *w2; bool vfirst;
        if ((c & 1) == 0) {
            int e = c >> 1; src = 4 * e + 1;
            if (e < 256) { w1 = wv1 + e * 3;  w2 = wh1 + e * 3;  vfirst = true;  }
            else { int q = e - 256; w1 = whb1 + q * 3; w2 = wvb1 + q * 3; vfirst = false; }
        } else {
            int e = (c - 1) >> 1; src = 4 * e + 3;
            if (e < 256) { w1 = wh2 + e * 3;  w2 = wv2 + e * 3;  vfirst = false; }
            else { int q = e - 256; w1 = wvb2 + q * 3; w2 = whb2 + q * 3; vfirst = true;  }
        }
        if (vfirst) {   // stage1 vertical(3x1), stage2 horizontal(1x3); stage2 taps valid cols only
#pragma unroll
            for (int dw = -1; dw <= 1; ++dw) {
                int ww2 = w + dw;
                if ((unsigned)ww2 >= 64u) continue;
                float mid = 0.f;
#pragma unroll
                for (int dh = -1; dh <= 1; ++dh) mid += w1[dh + 1] * ld(h + dh, ww2, src);
                out += w2[dw + 1] * mid;
            }
        } else {        // stage1 horizontal, stage2 vertical
#pragma unroll
            for (int dh = -1; dh <= 1; ++dh) {
                int hh2 = h + dh;
                if ((unsigned)hh2 >= 64u) continue;
                float mid = 0.f;
#pragma unroll
                for (int dw = -1; dw <= 1; ++dw) mid += w1[dw + 1] * ld(hh2, w + dw, src);
                out += w2[dh + 1] * mid;
            }
        }
    }
    xBCc[idx] = silu_f(out);
}

// ---------------------------------------------------------------- z branch: 3x3 dwconv + silu -> bf16 cat[:,1024:]
__global__ void zconv_kernel(const float* __restrict__ zx, const float* __restrict__ wz,
                             const float* __restrict__ alpha2, unsigned short* __restrict__ cat)
{
    long idx = (long)blockIdx.x * 256 + threadIdx.x;
    const long total = (long)BL * 1024;
    if (idx >= total) return;
    int ch  = (int)(idx & 1023);
    long bl = idx >> 10;
    int b = (int)(bl >> 12);
    int l = (int)(bl & 4095);
    int h = l >> 6, w = l & 63;
    const float* base = zx + (long)b * LL * D_IN_PROJ;
    const float* wk = wz + ch * 9;
    float out = 0.f;
#pragma unroll
    for (int kh = 0; kh < 3; ++kh) {
        int hh = h + kh - 1;
        if ((unsigned)hh >= 64u) continue;
#pragma unroll
        for (int kw = 0; kw < 3; ++kw) {
            int ww = w + kw - 1;
            if ((unsigned)ww >= 64u) continue;
            out += wk[kh * 3 + kw] * base[(long)(hh * 64 + ww) * D_IN_PROJ + ch];
        }
    }
    cat[bl * 2048 + 1024 + ch] = f2bf(silu_f(out) * alpha2[0]);
}

// ---------------------------------------------------------------- KV[b,m,g,s,p] = sum_l K[l,g,s] * x[l,2m+g,p]*dA[l,2m+g]
__global__ __launch_bounds__(256) void kv_kernel(const float* __restrict__ xBCc,
    const float* __restrict__ dA, float* __restrict__ KV)
{
    int b = blockIdx.x >> 1, g = blockIdx.x & 1;
    __shared__ float kbuf[32][16];
    int t = threadIdx.x;
    float acc[2][16] = {};
    int mq[2], pq[2], hd[2];
#pragma unroll
    for (int r = 0; r < 2; ++r) {
        int q = t + r * 256;            // q = m*8 + p, 512 (m,p) pairs
        mq[r] = q >> 3; pq[r] = q & 7; hd[r] = 2 * mq[r] + g;
    }
    const float* xb  = xBCc + (long)b * LL * CONV_DIM;
    const float* dab = dA   + (long)b * LL * 128;
    for (int l0 = 0; l0 < LL; l0 += 32) {
        __syncthreads();
#pragma unroll
        for (int r = 0; r < 2; ++r) {
            int ii = t + r * 256;
            int i = ii >> 4, s = ii & 15;
            kbuf[i][s] = xb[(long)(l0 + i) * CONV_DIM + 1024 + g * 16 + s];
        }
        __syncthreads();
        for (int i = 0; i < 32; ++i) {
            long row = (long)(l0 + i);
#pragma unroll
            for (int r = 0; r < 2; ++r) {
                float x = xb[row * CONV_DIM + hd[r] * 8 + pq[r]];
                float v = x * dab[row * 128 + hd[r]];
#pragma unroll
                for (int s = 0; s < 16; ++s) acc[r][s] += kbuf[i][s] * v;
            }
        }
    }
#pragma unroll
    for (int r = 0; r < 2; ++r)
        for (int s = 0; s < 16; ++s)
            KV[(long)b * 16384 + ((long)hd[r] * 16 + s) * 8 + pq[r]] = acc[r][s];
}

// ---------------------------------------------------------------- y = Q*KV + x*D, LayerNorm, scale -> bf16 cat[:, :1024]
__global__ __launch_bounds__(256) void y_ln_kernel(const float* __restrict__ xBCc,
    const float* __restrict__ KV, const float* __restrict__ Dp,
    const float* __restrict__ ln_g, const float* __restrict__ ln_b,
    const float* __restrict__ scale, const float* __restrict__ shift,
    const float* __restrict__ alpha1, unsigned short* __restrict__ cat)
{
    extern __shared__ float smem[];
    float* kvs = smem;            // 16384 floats: KV[b]
    float* qs  = smem + 16384;    // 32 floats: Q for this row
    float* red = smem + 16416;    // 512 floats: reduction
    long bl = blockIdx.x;
    int b = (int)(bl >> 12);
    int t = threadIdx.x;
    const float* kvb = KV + (long)b * 16384;
    for (int i = 0; i < 64; ++i) kvs[t + i * 256] = kvb[t + i * 256];
    const float* row = xBCc + bl * (long)CONV_DIM;
    if (t < 32) qs[t] = row[1056 + t];
    __syncthreads();

    float yv[4]; float s1 = 0.f, s2 = 0.f;
#pragma unroll
    for (int i = 0; i < 4; ++i) {
        int ch = t + i * 256;
        int head = ch >> 3, p = ch & 7;
        int g = head & 1;
        float x = row[ch];
        float acc = x * Dp[head];
        const float* kvp = kvs + head * 128 + p;   // (m*2+g)==head
        const float* qp  = qs + g * 16;
#pragma unroll
        for (int s = 0; s < 16; ++s) acc += qp[s] * kvp[s * 8];
        yv[i] = acc; s1 += acc; s2 += acc * acc;
    }
    red[t] = s1; red[t + 256] = s2;
    __syncthreads();
    for (int off = 128; off > 0; off >>= 1) {
        if (t < off) { red[t] += red[t + off]; red[t + 256] += red[t + 256 + off]; }
        __syncthreads();
    }
    float mu   = red[0]   * (1.f / 1024.f);
    float var  = red[256] * (1.f / 1024.f) - mu * mu;
    float rstd = rsqrtf(var + 1e-5f);
    float sc = scale[0], sh = shift[0], a1 = alpha1[0];
#pragma unroll
    for (int i = 0; i < 4; ++i) {
        int ch = t + i * 256;
        float yn = (yv[i] - mu) * rstd * ln_g[ch] + ln_b[ch];
        cat[bl * 2048 + ch] = f2bf(a1 * (yn * sc + sh));
    }
}

// ---------------------------------------------------------------- host
extern "C" void kernel_launch(void* const* d_in, const int* in_sizes, int n_in,
                              void* d_out, int out_size, void* d_ws, size_t ws_size,
                              hipStream_t stream) {
    (void)in_sizes; (void)n_in; (void)out_size; (void)ws_size;
    const float* u       = (const float*)d_in[0];
    const float* W_in    = (const float*)d_in[3];
    const float* dt_bias = (const float*)d_in[4];
    const float* A_log   = (const float*)d_in[5];
    const float* Dp      = (const float*)d_in[6];
    const float* w33     = (const float*)d_in[7];
    const float* w13x1   = (const float*)d_in[8];
    const float* w31x1   = (const float*)d_in[9];
    const float* w13x2   = (const float*)d_in[10];
    const float* w31x2   = (const float*)d_in[11];
    const float* w13bc1  = (const float*)d_in[12];
    const float* w31bc1  = (const float*)d_in[13];
    const float* w13bc2  = (const float*)d_in[14];
    const float* w31bc2  = (const float*)d_in[15];
    const float* wz      = (const float*)d_in[16];
    const float* ln_g    = (const float*)d_in[17];
    const float* ln_b    = (const float*)d_in[18];
    const float* scale   = (const float*)d_in[19];
    const float* shift   = (const float*)d_in[20];
    const float* alpha1  = (const float*)d_in[21];
    const float* alpha2  = (const float*)d_in[22];
    const float* W_out   = (const float*)d_in[23];
    float* out = (float*)d_out;

    char* ws = (char*)d_ws;
    size_t off = 0;
    auto wsalloc = [&](size_t bytes) -> char* {
        char* p = ws + off;
        off = (off + bytes + 255) & ~(size_t)255;
        return p;
    };
    unsigned short* u_bf    = (unsigned short*)wsalloc((size_t)BL * 512 * 2);
    unsigned short* win_bf  = (unsigned short*)wsalloc((size_t)2240 * 512 * 2);
    unsigned short* wout_bf = (unsigned short*)wsalloc((size_t)512 * 2048 * 2);
    float* zx    = (float*)wsalloc((size_t)BL * D_IN_PROJ * 4);
    float* xbcc  = (float*)wsalloc((size_t)BL * CONV_DIM * 4);
    float* dAbuf = (float*)wsalloc((size_t)BL * 128 * 4);
    float* kvbuf = (float*)wsalloc((size_t)8 * 16384 * 4);
    unsigned short* cat = (unsigned short*)wsalloc((size_t)BL * 2048 * 2);

    int n;
    n = BL * 512;    cvt_bf16<<<(n + 255) / 256, 256, 0, stream>>>(u, u_bf, n);
    n = 2240 * 512;  cvt_bf16<<<(n + 255) / 256, 256, 0, stream>>>(W_in, win_bf, n);
    n = 512 * 2048;  cvt_bf16<<<(n + 255) / 256, 256, 0, stream>>>(W_out, wout_bf, n);

    dim3 g1(BL / 256, D_IN_PROJ / 64);     // 128 x 35
    gemm_bf16_wmma<<<g1, 256, 0, stream>>>(u_bf, win_bf, zx, 512, D_IN_PROJ);

    n = BL * 128;
    dA_kernel<<<(n + 255) / 256, 256, 0, stream>>>(zx, dt_bias, A_log, dAbuf, n);

    long tot = (long)BL * CONV_DIM;
    conv_xbc_kernel<<<(unsigned)((tot + 255) / 256), 256, 0, stream>>>(
        zx, xbcc, w33, w31x1, w13x1, w13x2, w31x2, w13bc1, w31bc1, w13bc2, w31bc2);

    tot = (long)BL * 1024;
    zconv_kernel<<<(unsigned)((tot + 255) / 256), 256, 0, stream>>>(zx, wz, alpha2, cat);

    kv_kernel<<<16, 256, 0, stream>>>(xbcc, dAbuf, kvbuf);

    size_t smem = (size_t)(16384 + 32 + 512) * sizeof(float);
    y_ln_kernel<<<BL, 256, smem, stream>>>(xbcc, kvbuf, Dp, ln_g, ln_b, scale, shift, alpha1, cat);

    dim3 g2(BL / 256, 512 / 64);           // 128 x 8
    gemm_bf16_wmma<<<g2, 256, 0, stream>>>(cat, wout_bf, out, 2048, 512);
}
